// ConvScoreSSREM_68633577390676
// MI455X (gfx1250) — matrix operand order repaired and checked by reading, not compiled, not run on hardware
//
#include <hip/hip_runtime.h>
#include <hip/hip_bf16.h>
#include <math.h>

// ---------------------------------------------------------------------------
// ConvScoreSSREM on MI455X (gfx1250), wave32 + v_wmma_f32_16x16x32_bf16.
//   out[b] = scores[b,0] - logsumexp_k(scores[b,k]),
//   scores[b,k] = (contexts @ M)[b,:] . res_k[b,:]
// Fused: GEMM tile (split-bf16, fp32 accum) + per-row dots + log-softmax.
// ---------------------------------------------------------------------------

typedef __bf16  v16bf __attribute__((ext_vector_type(16)));
typedef __bf16  v8bf  __attribute__((ext_vector_type(8)));
typedef float   v8f   __attribute__((ext_vector_type(8)));

#define E 1024
#define APAD 8          // LDS row padding (keeps 16B alignment, avoids bank conflicts)

// ---------------------------------------------------------------------------
// Pre-pass: M[k][j] fp32 -> transposed split-bf16 Mt_hi/Mt_lo[j][k].
// Transposed layout makes each WMMA B-fragment a contiguous 32B load per lane.
// ---------------------------------------------------------------------------
__global__ void prep_M_kernel(const float* __restrict__ M,
                              __bf16* __restrict__ Mt_hi,
                              __bf16* __restrict__ Mt_lo) {
    __shared__ __bf16 th[32][33];
    __shared__ __bf16 tl[32][33];
    const int j0 = blockIdx.x * 32;
    const int k0 = blockIdx.y * 32;
    const int tx = threadIdx.x;      // 0..31
    const int ty = threadIdx.y;      // 0..7
#pragma unroll
    for (int i = 0; i < 4; ++i) {
        int k = k0 + ty + 8 * i;
        float x = M[(size_t)k * E + j0 + tx];          // coalesced 128B rows
        __bf16 h = (__bf16)x;
        th[ty + 8 * i][tx] = h;
        tl[ty + 8 * i][tx] = (__bf16)(x - (float)h);
    }
    __syncthreads();
#pragma unroll
    for (int i = 0; i < 4; ++i) {
        int j = j0 + ty + 8 * i;
        Mt_hi[(size_t)j * E + k0 + tx] = th[tx][ty + 8 * i];
        Mt_lo[(size_t)j * E + k0 + tx] = tl[tx][ty + 8 * i];
    }
}

// ---------------------------------------------------------------------------
// Main fused kernel. 1 block = 16 rows of B, 256 threads = 8 wave32 waves.
// Wave w handles N-chunks j0 = 16*w + 128*t (t = 0..7).
// ---------------------------------------------------------------------------
__global__ void __launch_bounds__(256)
conv_score_kernel(const float* __restrict__ contexts,
                  const float* __restrict__ res0,
                  const float* __restrict__ res1,
                  const float* __restrict__ res2,
                  const float* __restrict__ res3,
                  const float* __restrict__ res4,
                  const __bf16* __restrict__ Mt_hi,
                  const __bf16* __restrict__ Mt_lo,
                  float* __restrict__ out) {
    __shared__ __bf16 As_hi[16][E + APAD];
    __shared__ __bf16 As_lo[16][E + APAD];
    __shared__ float  scores_s[8][16][5];   // per-wave partial scores

    const int tid  = threadIdx.x;
    const int lane = tid & 31;
    const int wv   = tid >> 5;              // wave id 0..7
    const int r0   = blockIdx.x * 16;       // first row of this block

    // --- Stage contexts rows into LDS as split bf16 (each element converted once) ---
    for (int i = tid; i < 16 * E; i += 256) {
        int r = i >> 10, c = i & (E - 1);
        float x = contexts[(size_t)(r0 + r) * E + c];   // coalesced
        __bf16 h = (__bf16)x;
        As_hi[r][c] = h;
        As_lo[r][c] = (__bf16)(x - (float)h);
    }
    __syncthreads();

    const float* resp[5] = {res0, res1, res2, res3, res4};

    const int arow  = lane & 15;            // A row / B column / C column within tile
    const int apart = lane >> 4;            // half-wave selector
    float sacc[5][8];
#pragma unroll
    for (int k = 0; k < 5; ++k)
#pragma unroll
        for (int v = 0; v < 8; ++v) sacc[k][v] = 0.0f;

    for (int j0 = wv * 16; j0 < E; j0 += 128) {
        // Prefetch next res chunk for this wave (global_prefetch_b8 path)
        if (j0 + 128 < E) {
#pragma unroll
            for (int k = 0; k < 5; ++k)
                __builtin_prefetch(&resp[k][(size_t)(r0 + arow) * E + j0 + 128], 0, 1);
        }

        v8f acc = {};   // fp32 accumulator: cm[16 x 16] tile
#pragma unroll 4
        for (int k0 = 0; k0 < E; k0 += 32) {
            // A fragments (16x32 bf16) from LDS: two 16B chunks per lane
            v8bf a0h = *(const v8bf*)&As_hi[arow][k0 + 16 * apart];
            v8bf a1h = *(const v8bf*)&As_hi[arow][k0 + 16 * apart + 8];
            v8bf a0l = *(const v8bf*)&As_lo[arow][k0 + 16 * apart];
            v8bf a1l = *(const v8bf*)&As_lo[arow][k0 + 16 * apart + 8];
            v16bf a_hi = __builtin_shufflevector(a0h, a1h,
                0,1,2,3,4,5,6,7,8,9,10,11,12,13,14,15);
            v16bf a_lo = __builtin_shufflevector(a0l, a1l,
                0,1,2,3,4,5,6,7,8,9,10,11,12,13,14,15);

            // B fragments (32x16 bf16) from transposed M: contiguous 32B per lane
            const size_t boff = (size_t)(j0 + arow) * E + k0 + 16 * apart;
            v16bf b_hi = *(const v16bf*)(Mt_hi + boff);
            v16bf b_lo = *(const v16bf*)(Mt_lo + boff);

            // Split-bf16 fp32 emulation: hi*hi + hi*lo + lo*hi (fp32 accumulate)
            acc = __builtin_amdgcn_wmma_f32_16x16x32_bf16(
                false, a_hi, false, b_hi, (short)0, acc, false, false);
            acc = __builtin_amdgcn_wmma_f32_16x16x32_bf16(
                false, a_hi, false, b_lo, (short)0, acc, false, false);
            acc = __builtin_amdgcn_wmma_f32_16x16x32_bf16(
                false, a_lo, false, b_hi, (short)0, acc, false, false);
        }

        // --- Fused score epilogue on the C layout:
        //     lane = column, VGPR v = row v (lanes 0-15) / v+8 (lanes 16-31).
        const int col = j0 + arow;
#pragma unroll
        for (int k = 0; k < 5; ++k) {
            const float* rk = resp[k];
#pragma unroll
            for (int v = 0; v < 8; ++v) {
                int row = r0 + v + 8 * apart;
                float x = acc[v] * rk[(size_t)row * E + col];  // coalesced per half-wave
                x += __shfl_xor(x, 8, 32);   // reduce 16 columns within each half
                x += __shfl_xor(x, 4, 32);
                x += __shfl_xor(x, 2, 32);
                x += __shfl_xor(x, 1, 32);
                sacc[k][v] += x;
            }
        }
    }

    // --- Deterministic cross-wave score reduction via LDS (fixed order, no FP atomics)
#pragma unroll
    for (int k = 0; k < 5; ++k)
#pragma unroll
        for (int v = 0; v < 8; ++v)
            if ((lane & 15) == k)
                scores_s[wv][v + 8 * apart][k] = sacc[k][v];
    __syncthreads();

    if (tid < 16) {
        float s[5];
#pragma unroll
        for (int k = 0; k < 5; ++k) {
            float acc_s = 0.0f;
#pragma unroll
            for (int w = 0; w < 8; ++w) acc_s += scores_s[w][tid][k];
            s[k] = acc_s;
        }
        float m = s[0];
#pragma unroll
        for (int k = 1; k < 5; ++k) m = fmaxf(m, s[k]);
        float se = 0.0f;
#pragma unroll
        for (int k = 0; k < 5; ++k) se += expf(s[k] - m);
        out[r0 + tid] = s[0] - (m + logf(se));
    }
}

// ---------------------------------------------------------------------------
// Host launcher
// ---------------------------------------------------------------------------
extern "C" void kernel_launch(void* const* d_in, const int* in_sizes, int n_in,
                              void* d_out, int out_size, void* d_ws, size_t ws_size,
                              hipStream_t stream) {
    const float* contexts = (const float*)d_in[0];
    const float* res_pos  = (const float*)d_in[1];
    const float* res_neg1 = (const float*)d_in[2];
    const float* res_neg2 = (const float*)d_in[3];
    const float* res_neg3 = (const float*)d_in[4];
    const float* res_neg4 = (const float*)d_in[5];
    const float* mat_M    = (const float*)d_in[6];
    float* out = (float*)d_out;

    const int B = in_sizes[0] / E;          // 16384

    __bf16* Mt_hi = (__bf16*)d_ws;          // 1024*1024 bf16 = 2 MB
    __bf16* Mt_lo = Mt_hi + (size_t)E * E;  // +2 MB (ws must be >= 4 MB)

    dim3 pgrid(E / 32, E / 32);
    dim3 pblk(32, 8);
    prep_M_kernel<<<pgrid, pblk, 0, stream>>>(mat_M, Mt_hi, Mt_lo);

    conv_score_kernel<<<B / 16, 256, 0, stream>>>(
        contexts, res_pos, res_neg1, res_neg2, res_neg3, res_neg4,
        Mt_hi, Mt_lo, out);
}
